// Attention_69887707840840
// MI455X (gfx1250) — matrix operand order repaired and checked by reading, not compiled, not run on hardware
//
#include <hip/hip_runtime.h>

// ---------------- shapes (fixed) ----------------
// B=4 S=1024 P=1024 E=1024 H=16 D=64 ns=2048
#define SB 4
#define SS 1024
#define SP 1024
#define SE 1024
#define SH 16
#define SD 64
#define SNS 2048

typedef __attribute__((ext_vector_type(16))) __bf16 v16bf;
typedef __attribute__((ext_vector_type(8)))  __bf16 v8bf;
typedef __attribute__((ext_vector_type(8)))  float  v8f;
typedef __attribute__((ext_vector_type(4)))  float  f32x4;
typedef __attribute__((ext_vector_type(4)))  unsigned int u32x4;
typedef __attribute__((ext_vector_type(8)))  unsigned int u32x8;

union ABu { v16bf v; v8bf h[2]; };

__device__ __forceinline__ v8f wmma_bf16(v16bf a, v16bf b, v8f c) {
  // D = A(16x32) * B(32x16) + C, f32 accumulate
  return __builtin_amdgcn_wmma_f32_16x16x32_bf16(false, a, false, b,
                                                 (short)0, c, false, false);
}

// CDNA5 async copy: global -> LDS, 16 bytes, tracked by ASYNCcnt.
__device__ __forceinline__ void async_ld_b128(unsigned lds_off, const void* g) {
  asm volatile("global_load_async_to_lds_b128 %0, %1, off"
               :: "v"(lds_off), "v"((unsigned long long)(uintptr_t)g)
               : "memory");
}
__device__ __forceinline__ void wait_async0() {
  asm volatile("s_wait_asynccnt 0x0" ::: "memory");
}
__device__ __forceinline__ unsigned lds_off_u32(const void* p) {
  // LDS aperture flat address: low 32 bits are the LDS byte offset.
  return (unsigned)(uintptr_t)p;
}

// CDNA5 Tensor Data Mover: 2-D tile of bf16 (data_size=2B), global -> LDS.
// Descriptor per CDNA5 ISA (D# group0 + group1).
__device__ __forceinline__ void tdm_load_2d(unsigned lds_addr, const void* gptr,
                                            unsigned tensor_d0, unsigned tensor_d1,
                                            unsigned tile_d0, unsigned tile_d1,
                                            unsigned stride0) {
  unsigned long long ga = (unsigned long long)(uintptr_t)gptr;
  u32x4 g0;
  g0[0] = 1u;                                   // count=1, user descriptor
  g0[1] = lds_addr;                             // lds_addr [63:32]
  g0[2] = (unsigned)(ga & 0xffffffffu);         // global_addr [95:64]
  g0[3] = (unsigned)((ga >> 32) & 0x01ffffffu)  // global_addr [120:96]
          | 0x80000000u;                        // type=2 ("image") [127:126]
  u32x8 g1;
  g1[0] = 0x00010000u;                                   // data_size=1 (2B)
  g1[1] = (tensor_d0 & 0xffffu) << 16;                   // tensor_dim0 lo16
  g1[2] = ((tensor_d0 >> 16) & 0xffffu)                  // tensor_dim0 hi16
          | ((tensor_d1 & 0xffffu) << 16);               // tensor_dim1 lo16
  g1[3] = ((tensor_d1 >> 16) & 0xffffu)                  // tensor_dim1 hi16
          | ((tile_d0 & 0xffffu) << 16);                 // tile_dim0
  g1[4] = tile_d1 & 0xffffu;                             // tile_dim1, tile_dim2=0
  g1[5] = stride0;                                       // tensor_dim0_stride lo32
  g1[6] = 0u;                                            // stride0 hi16 | stride1 lo16
  g1[7] = 0u;
  asm volatile("tensor_load_to_lds %0, %1" :: "s"(g0), "s"(g1) : "memory");
}

// ---------------- f32 -> bf16 convert (8 elems/thread) ----------------
__global__ __launch_bounds__(256) void cvt_bf16(const float* __restrict__ src,
                                                __bf16* __restrict__ dst,
                                                size_t n8) {  // n/8
  size_t i = (size_t)blockIdx.x * blockDim.x + threadIdx.x;
  if (i >= n8) return;
  f32x4 a = ((const f32x4*)src)[2 * i];
  f32x4 b = ((const f32x4*)src)[2 * i + 1];
  v8bf o;
  o[0] = (__bf16)a[0]; o[1] = (__bf16)a[1]; o[2] = (__bf16)a[2]; o[3] = (__bf16)a[3];
  o[4] = (__bf16)b[0]; o[5] = (__bf16)b[1]; o[6] = (__bf16)b[2]; o[7] = (__bf16)b[3];
  ((v8bf*)dst)[i] = o;
}

// ---------------- layer_past -> present(f32) + bf16 kv cache ----------------
__global__ __launch_bounds__(256) void past_copy(const float* __restrict__ past,
                                                 float* __restrict__ presK,
                                                 float* __restrict__ presV,
                                                 __bf16* __restrict__ kbw,
                                                 __bf16* __restrict__ vbw) {
  const size_t NTOT4 = (size_t)SB * SH * SP * SD / 4;  // 1,048,576 quads
  size_t i = (size_t)blockIdx.x * blockDim.x + threadIdx.x;
  if (i >= NTOT4) return;
  int d4 = (int)(i & 15);                 // quad within D
  size_t t = i >> 4;
  int p = (int)(t & (SP - 1)); t >>= 10;
  int h = (int)(t & (SH - 1));
  int b = (int)(t >> 4);
  f32x4 kv = ((const f32x4*)past)[i];
  f32x4 vv = ((const f32x4*)past)[i + NTOT4];
  size_t dst = (((((size_t)b * SH + h) * SNS) + p) * SD) / 4 + d4;
  ((f32x4*)presK)[dst] = kv;
  ((f32x4*)presV)[dst] = vv;
  __bf16* kd = kbw + dst * 4;
  __bf16* vd = vbw + dst * 4;
  kd[0] = (__bf16)kv[0]; kd[1] = (__bf16)kv[1]; kd[2] = (__bf16)kv[2]; kd[3] = (__bf16)kv[3];
  vd[0] = (__bf16)vv[0]; vd[1] = (__bf16)vv[1]; vd[2] = (__bf16)vv[2]; vd[3] = (__bf16)vv[3];
}

// ---------------- tiled bf16 GEMM, 128x128x64, 8 waves, double-buffered ----
// A tile staged by the Tensor Data Mover; B tile transposed by VALU.
// MODE 0: qkv epilogue (scatter q/k/v).  MODE 1: f32 out + bias.
template <int MODE>
__global__ __launch_bounds__(256) void gemm_bf16(
    const __bf16* __restrict__ A, const __bf16* __restrict__ Bw,
    const float* __restrict__ bias, int N, int K,
    float* __restrict__ outF, __bf16* __restrict__ qbw,
    float* __restrict__ presK, float* __restrict__ presV,
    __bf16* __restrict__ kbw, __bf16* __restrict__ vbw) {
  __shared__ __align__(64) __bf16 As[2][128][64];  // row-major (M x K)  2x16KB
  __shared__ __align__(64) __bf16 Bs[2][128][64];  // N-major  (N x K)   2x16KB

  const int tid = threadIdx.x;
  const int wave = tid >> 5, lane = tid & 31;
  const int l16 = lane & 15, hif = lane >> 4;
  const int bm = blockIdx.y * 128;
  const int bn = blockIdx.x * 128;
  const int wm = (wave >> 2) * 64;  // 0 or 64
  const int wn = (wave & 3) * 32;   // 0..96

  v8f acc[4][2];
#pragma unroll
  for (int i = 0; i < 4; ++i)
#pragma unroll
    for (int j = 0; j < 2; ++j) acc[i][j] = (v8f)0.0f;

  auto stageA = [&](int k0, int buf) {   // TDM: one DMA per tile, wave 0 only
    if (wave == 0) {
      const __bf16* ga = A + (size_t)bm * K + k0;
      tdm_load_2d(lds_off_u32(&As[buf][0][0]), ga,
                  /*tensor_d0=*/64u, /*tensor_d1=*/128u,
                  /*tile_d0=*/64u, /*tile_d1=*/128u, /*stride0=*/(unsigned)K);
    }
  };
  auto stageB = [&](int k0, int buf) {   // transpose [K,N] rows into Bs[N][K]
    int kk = tid >> 2;            // 0..63
    int c = (tid & 3) * 32;       // 0..96
    const __bf16* src = Bw + (size_t)(k0 + kk) * N + bn + c;
    v8bf bb[4];
    bb[0] = *(const v8bf*)(src);
    bb[1] = *(const v8bf*)(src + 8);
    bb[2] = *(const v8bf*)(src + 16);
    bb[3] = *(const v8bf*)(src + 24);
#pragma unroll
    for (int t = 0; t < 32; ++t) Bs[buf][c + t][kk] = bb[t >> 3][t & 7];
  };

  // prologue: fill buffer 0
  stageA(0, 0);
  stageB(0, 0);
  if (wave == 0) __builtin_amdgcn_s_wait_tensorcnt(0);
  __syncthreads();

  for (int k0 = 0; k0 < K; k0 += 64) {
    const int cur = (k0 >> 6) & 1, nxt = cur ^ 1;
    if (k0 + 64 < K) {           // overlap next stage with compute
      stageA(k0 + 64, nxt);
      stageB(k0 + 64, nxt);
    }
#pragma unroll
    for (int ks = 0; ks < 64; ks += 32) {
      // A fragments: 16x32, lane<16 K 0..15, lane>=16 K 16..31
      v16bf af[4];
#pragma unroll
      for (int i = 0; i < 4; ++i) {
        int r = wm + i * 16 + l16;
        int kk = ks + hif * 16;
        ABu u;
        u.h[0] = *(const v8bf*)&As[cur][r][kk];
        u.h[1] = *(const v8bf*)&As[cur][r][kk + 8];
        af[i] = u.v;
      }
      // B fragments: lane<16 K 0..15, lane>=16 K 16..31 (N-major LDS)
      v16bf bfr[2];
#pragma unroll
      for (int j = 0; j < 2; ++j) {
        int n = wn + j * 16 + l16;
        bfr[j] = *(const v16bf*)&Bs[cur][n][ks + hif * 16];
      }
#pragma unroll
      for (int i = 0; i < 4; ++i)
#pragma unroll
        for (int j = 0; j < 2; ++j)
          acc[i][j] = wmma_bf16(af[i], bfr[j], acc[i][j]);
    }
    if (wave == 0) __builtin_amdgcn_s_wait_tensorcnt(0);
    __syncthreads();
  }

  // epilogue
#pragma unroll
  for (int i = 0; i < 4; ++i)
#pragma unroll
    for (int j = 0; j < 2; ++j)
#pragma unroll
      for (int e = 0; e < 8; ++e) {
        int m = bm + wm + i * 16 + e + hif * 8;   // global row (b*1024+s)
        int c = bn + wn + j * 16 + l16;           // global col
        float v = acc[i][j][e] + bias[c];
        if (MODE == 0) {
          int which = c >> 10, ee = c & 1023;
          int h = ee >> 6, d = ee & 63;
          int b = m >> 10, s = m & 1023;
          if (which == 0) {
            qbw[((((size_t)b * SH + h) * SS) + s) * SD + d] = (__bf16)v;
          } else {
            size_t idx = ((((size_t)b * SH + h) * SNS) + SP + s) * SD + d;
            if (which == 1) { presK[idx] = v; kbw[idx] = (__bf16)v; }
            else            { presV[idx] = v; vbw[idx] = (__bf16)v; }
          }
        } else {
          outF[(size_t)m * N + c] = v;
        }
      }
}

// ---------------- flash attention: 4 waves/block, 16 queries/wave ----------------
__global__ __launch_bounds__(128) void attn_kernel(const __bf16* __restrict__ qb,
                                                   const __bf16* __restrict__ kb,
                                                   const __bf16* __restrict__ vb,
                                                   __bf16* __restrict__ ab) {
  __shared__ __align__(64) __bf16 Ks[32][64];     // key-major (d contiguous)
  __shared__ __align__(64) __bf16 Vt[64][32];     // d-major   (key contiguous)
  __shared__ __align__(64) __bf16 Ps[4][16][32];  // prob tile per wave

  const int tid = threadIdx.x;
  const int wave = tid >> 5, lane = tid & 31;
  const int l16 = lane & 15, hif = lane >> 4;
  const int bh = blockIdx.y;            // b*16+h
  const int q0 = blockIdx.x * 64;       // block query base
  const int qw = q0 + wave * 16;        // wave query base

  const __bf16* qp = qb + (size_t)bh * SS * SD;
  const __bf16* kp = kb + (size_t)bh * SNS * SD;
  const __bf16* vp = vb + (size_t)bh * SNS * SD;

  // Q A-fragments: 16x64 -> two 16x32
  v16bf aq[2];
  {
    int r = qw + l16;
#pragma unroll
    for (int i = 0; i < 2; ++i) {
      int k0 = i * 32 + hif * 16;
      ABu u;
      u.h[0] = *(const v8bf*)(qp + (size_t)r * SD + k0);
      u.h[1] = *(const v8bf*)(qp + (size_t)r * SD + k0 + 8);
      aq[i] = u.v;
    }
  }

  v8f acc[4];
  float mrow[8], lrow[8];
#pragma unroll
  for (int t = 0; t < 4; ++t) acc[t] = (v8f)0.0f;
#pragma unroll
  for (int e = 0; e < 8; ++e) { mrow[e] = -3.0e38f; lrow[e] = 0.0f; }

  const int ntile = (SP + q0 + 64) >> 5;  // 32 keys per iteration
  for (int jt = 0; jt < ntile; ++jt) {
    const int kbase = jt * 32;
    // stage K via async-to-LDS; V transposed via manual scatter
    {
      int r = tid >> 2;            // 0..31 key
      int c = (tid & 3) * 16;      // d chunk
      const __bf16* ksrc = kp + (size_t)(kbase + r) * SD + c;
      unsigned lo = lds_off_u32(&Ks[r][c]);
      async_ld_b128(lo, ksrc);
      async_ld_b128(lo + 16, ksrc + 8);
      ABu u;
      const __bf16* vsrc = vp + (size_t)(kbase + r) * SD + c;
      u.h[0] = *(const v8bf*)(vsrc);
      u.h[1] = *(const v8bf*)(vsrc + 8);
#pragma unroll
      for (int t = 0; t < 16; ++t) Vt[c + t][r] = u.v[t];
      if (jt + 1 < ntile) {        // global_prefetch_b8 for next V tile
        __builtin_prefetch(vp + (size_t)(kbase + 32 + r) * SD + c, 0, 1);
      }
    }
    wait_async0();
    __syncthreads();

    // scores: two 16(q)x16(key) tiles, K-dim = d (two 32-chunks chained)
    v8f s[2];
#pragma unroll
    for (int t = 0; t < 2; ++t) {
      int n = t * 16 + l16;  // key within tile pair
      v16bf b0 = *(const v16bf*)&Ks[n][0 + hif * 16];
      v16bf b1 = *(const v16bf*)&Ks[n][32 + hif * 16];
      v8f z = (v8f)0.0f;
      s[t] = wmma_bf16(aq[0], b0, z);
      s[t] = wmma_bf16(aq[1], b1, s[t]);
    }

    // scale + causal mask + online softmax (rows live in 16-lane groups)
    float mnew[8];
#pragma unroll
    for (int e = 0; e < 8; ++e) {
      int qrow = qw + e + hif * 8;
      float s0 = s[0][e] * 0.125f;
      float s1 = s[1][e] * 0.125f;
      if (kbase + l16 > SP + qrow) s0 = -10000.0f;
      if (kbase + 16 + l16 > SP + qrow) s1 = -10000.0f;
      s[0][e] = s0; s[1][e] = s1;
      float mx = fmaxf(s0, s1);
#pragma unroll
      for (int d = 1; d < 16; d <<= 1) mx = fmaxf(mx, __shfl_xor(mx, d, 16));
      mnew[e] = fmaxf(mrow[e], mx);
    }
#pragma unroll
    for (int e = 0; e < 8; ++e) {
      float alpha = __expf(mrow[e] - mnew[e]);
      float p0 = __expf(s[0][e] - mnew[e]);
      float p1 = __expf(s[1][e] - mnew[e]);
      float rs = p0 + p1;
#pragma unroll
      for (int d = 1; d < 16; d <<= 1) rs += __shfl_xor(rs, d, 16);
      lrow[e] = lrow[e] * alpha + rs;
      mrow[e] = mnew[e];
#pragma unroll
      for (int t = 0; t < 4; ++t) acc[t][e] *= alpha;
      int pr = e + hif * 8;                        // C-layout -> LDS row-major
      Ps[wave][pr][l16] = (__bf16)p0;
      Ps[wave][pr][l16 + 16] = (__bf16)p1;
    }

    // reload P as A-fragment (16x32)
    v16bf ap;
    {
      int k0 = hif * 16;
      ABu u;
      u.h[0] = *(const v8bf*)&Ps[wave][l16][k0];
      u.h[1] = *(const v8bf*)&Ps[wave][l16][k0 + 8];
      ap = u.v;
    }
    // acc += P(16x32) * V(32x16) for each of 4 d-tiles
#pragma unroll
    for (int dt = 0; dt < 4; ++dt) {
      v16bf bv = *(const v16bf*)&Vt[dt * 16 + l16][hif * 16];
      acc[dt] = wmma_bf16(ap, bv, acc[dt]);
    }
    __syncthreads();
  }

  // normalize + write merged-head activation (bf16 for proj GEMM)
  const int b = bh >> 4, h = bh & 15;
#pragma unroll
  for (int e = 0; e < 8; ++e) {
    int srow = qw + e + hif * 8;
    float inv = 1.0f / lrow[e];
#pragma unroll
    for (int dt = 0; dt < 4; ++dt) {
      int col = h * SD + dt * 16 + l16;
      ab[(((size_t)b * SS) + srow) * SE + col] = (__bf16)(acc[dt][e] * inv);
    }
  }
}

// ---------------- launcher ----------------
extern "C" void kernel_launch(void* const* d_in, const int* in_sizes, int n_in,
                              void* d_out, int out_size, void* d_ws, size_t ws_size,
                              hipStream_t stream) {
  const float* x      = (const float*)d_in[0];
  const float* past   = (const float*)d_in[1];
  const float* w_attn = (const float*)d_in[2];
  const float* b_attn = (const float*)d_in[3];
  const float* w_proj = (const float*)d_in[4];
  const float* b_proj = (const float*)d_in[5];

  float* out = (float*)d_out;                            // [4,1024,1024]
  float* presK = out + (size_t)SB * SS * SE;             // [4,16,2048,64]
  float* presV = presK + (size_t)SB * SH * SNS * SD;

  char* ws = (char*)d_ws;
  __bf16* xb  = (__bf16*)(ws);                           //  8 MB  x bf16
  __bf16* wab = (__bf16*)(ws + (size_t)(8) * 1048576);   //  6 MB  w_attn bf16
  __bf16* wpb = (__bf16*)(ws + (size_t)(14) * 1048576);  //  2 MB  w_proj bf16
  __bf16* qbw = (__bf16*)(ws + (size_t)(16) * 1048576);  //  8 MB  q [B,H,S,D]
  __bf16* kbw = (__bf16*)(ws + (size_t)(24) * 1048576);  // 16 MB  k [B,H,ns,D]
  __bf16* vbw = (__bf16*)(ws + (size_t)(40) * 1048576);  // 16 MB  v [B,H,ns,D]
  __bf16* abw = (__bf16*)(ws + (size_t)(56) * 1048576);  //  8 MB  attn out bf16

  const size_t nx = (size_t)SB * SS * SE;        // 4,194,304
  const size_t nwa = (size_t)SE * 3 * SE;        // 3,145,728
  const size_t nwp = (size_t)SE * SE;            // 1,048,576

  cvt_bf16<<<dim3((unsigned)(nx / 8 / 256)), 256, 0, stream>>>(x, xb, nx / 8);
  cvt_bf16<<<dim3((unsigned)(nwa / 8 / 256)), 256, 0, stream>>>(w_attn, wab, nwa / 8);
  cvt_bf16<<<dim3((unsigned)(nwp / 8 / 256)), 256, 0, stream>>>(w_proj, wpb, nwp / 8);

  past_copy<<<dim3((unsigned)(nx / 4 / 256)), 256, 0, stream>>>(
      past, presK, presV, kbw, vbw);

  // qkv = x @ w_attn + b_attn  (M=4096, N=3072, K=1024)
  gemm_bf16<0><<<dim3(3 * SE / 128, SB * SS / 128), 256, 0, stream>>>(
      xb, wab, b_attn, 3 * SE, SE, nullptr, qbw, presK, presV, kbw, vbw);

  // flash attention
  attn_kernel<<<dim3(SS / 64, SB * SH), 128, 0, stream>>>(qbw, kbw, vbw, abw);

  // out = a @ w_proj + b_proj  (M=4096, N=1024, K=1024)
  gemm_bf16<1><<<dim3(SE / 128, SB * SS / 128), 256, 0, stream>>>(
      abw, wpb, b_proj, SE, SE, out, nullptr, nullptr, nullptr, nullptr, nullptr);
}
